// RNNModel_85289460564030
// MI455X (gfx1250) — compile-verified, matrix operand order checked by reading, not run on hardware
//
#include <hip/hip_runtime.h>
#include <hip/hip_bf16.h>

// ---------------------------------------------------------------------------
// Problem constants (from reference): V=50000 E=300 H=150 B=64 T=512 ATTN=75
// Padded: E->320, H->160, 3H(450)->480
// ---------------------------------------------------------------------------
#define BB 64
#define TT 512
#define EE 300
#define HH 150
#define H3 450
#define EPAD 320
#define HPAD 160
#define NPAD 480
#define ROWS (TT * BB) // 32768

typedef __attribute__((ext_vector_type(8)))  float  v8f;
typedef __attribute__((ext_vector_type(16))) __bf16 v16bf;

__device__ __forceinline__ unsigned short f32bf(float f) {
    unsigned int u = __float_as_uint(f);
    unsigned int r = u + 0x7FFFu + ((u >> 16) & 1u);
    return (unsigned short)(r >> 16);
}
__device__ __forceinline__ float bf16f(unsigned short h) {
    return __uint_as_float(((unsigned int)h) << 16);
}
__device__ __forceinline__ float sigmoidf_(float x) {
    return 1.0f / (1.0f + __expf(-x));
}
__device__ __forceinline__ v8f wmma_bf16(v16bf a, v16bf b, v8f c) {
    return __builtin_amdgcn_wmma_f32_16x16x32_bf16(
        false, a, false, b, (short)0, c, false, false);
}

// ---------------------------------------------------------------------------
// Weight pad+convert: src fp32 [rows,cols] -> dst bf16 [Rpad,Kpad], zero pad
// ---------------------------------------------------------------------------
__global__ void pad_convert(const float* __restrict__ src, int rows, int cols,
                            unsigned short* __restrict__ dst, int Rpad, int Kpad) {
    int idx = blockIdx.x * blockDim.x + threadIdx.x;
    int total = Rpad * Kpad;
    if (idx >= total) return;
    int r = idx / Kpad, k = idx - r * Kpad;
    float v = (r < rows && k < cols) ? src[(size_t)r * cols + k] : 0.0f;
    dst[idx] = f32bf(v);
}

// ---------------------------------------------------------------------------
// Embedding gather -> bf16 activations x0 [ROWS, EPAD], row = t*B + b
// texts is [B, T] int32
// ---------------------------------------------------------------------------
__global__ void gather_embed(const int* __restrict__ texts,
                             const float* __restrict__ emb,
                             unsigned short* __restrict__ x0) {
    size_t idx = (size_t)blockIdx.x * blockDim.x + threadIdx.x;
    size_t total = (size_t)ROWS * EPAD;
    if (idx >= total) return;
    int row = (int)(idx / EPAD);
    int k   = (int)(idx - (size_t)row * EPAD);
    int t = row >> 6, b = row & 63;
    float v = 0.0f;
    if (k < EE) {
        int tok = texts[b * TT + t];
        v = emb[(size_t)tok * EE + k];
    }
    x0[idx] = f32bf(v);
}

// ---------------------------------------------------------------------------
// WMMA GEMM: C[M,N] = A[M,Kpad] (bf16, row-major) x Bw[N,Kpad]^T (bf16) + bias
// 2x2 block of 16x16 tiles per wave: 2 A-frags + 2 B-frags feed 4 WMMAs per
// K-chunk (operand reuse x2 on both sides). Fragment layout used here (contiguous k, not the ISA table order),
// 16-bit A/B: lane 0-15 row (lane&15) K=[0,16), lane 16-31 K=[16,32);
// 16 contiguous bf16 (32 bytes) per lane per fragment.
// Requires M % 32 == 0 and N % 32 == 0.
// ---------------------------------------------------------------------------
__global__ __launch_bounds__(256)
void wmma_gemm_bf16(const unsigned short* __restrict__ A,
                    const unsigned short* __restrict__ Bw,
                    const float* __restrict__ bias, int biasLen,
                    float* __restrict__ C,
                    int M, int Kpad, int N) {
    int lane = threadIdx.x & 31;
    int wave = threadIdx.x >> 5;
    int nJobs = N >> 5;                 // N tile-pairs
    int job = blockIdx.x * (blockDim.x >> 5) + wave;
    if (job >= (M >> 5) * nJobs) return;
    int mj = job / nJobs, nj = job - mj * nJobs;
    int m0 = mj << 5, n0 = nj << 5;
    int krow = (lane >> 4) << 4;        // 0 or 16
    const unsigned short* a0 = A  + (size_t)(m0 + (lane & 15)) * Kpad;
    const unsigned short* a1 = a0 + (size_t)16 * Kpad;
    const unsigned short* b0 = Bw + (size_t)(n0 + (lane & 15)) * Kpad;
    const unsigned short* b1 = b0 + (size_t)16 * Kpad;
    v8f acc00 = {}, acc01 = {}, acc10 = {}, acc11 = {};
    for (int k = 0; k < Kpad; k += 32) {
        v16bf av0 = *reinterpret_cast<const v16bf*>(a0 + k + krow);
        v16bf av1 = *reinterpret_cast<const v16bf*>(a1 + k + krow);
        v16bf bv0 = *reinterpret_cast<const v16bf*>(b0 + k + krow);
        v16bf bv1 = *reinterpret_cast<const v16bf*>(b1 + k + krow);
        acc00 = wmma_bf16(av0, bv0, acc00);
        acc01 = wmma_bf16(av0, bv1, acc01);
        acc10 = wmma_bf16(av1, bv0, acc10);
        acc11 = wmma_bf16(av1, bv1, acc11);
    }
    int nA = n0 + (lane & 15);
    int nB = nA + 16;
    float bvA = (bias != nullptr && nA < biasLen) ? bias[nA] : 0.0f;
    float bvB = (bias != nullptr && nB < biasLen) ? bias[nB] : 0.0f;
    int mb0 = m0 + ((lane >> 4) << 3);
    int mb1 = mb0 + 16;
    for (int i = 0; i < 8; ++i) {
        C[(size_t)(mb0 + i) * N + nA] = acc00[i] + bvA;
        C[(size_t)(mb0 + i) * N + nB] = acc01[i] + bvB;
        C[(size_t)(mb1 + i) * N + nA] = acc10[i] + bvA;
        C[(size_t)(mb1 + i) * N + nB] = acc11[i] + bvB;
    }
}

// ---------------------------------------------------------------------------
// Persistent GRU layer: single workgroup (512 thr = 16 waves) on one WGP.
// LDS (dynamic, 296,960 B):
//   sWhh : bf16 [NPAD][HPAD]   = 153600 B  (recurrent weights, resident)
//   sHg  : f32  [BB][NPAD]     = 122880 B  (gate pre-activations)
//   sH   : bf16 [BB][HPAD]     =  20480 B  (hidden state)
// Whh preload via ASYNCcnt-tracked global->LDS b128 copies when the toolchain
// exposes the gfx1250 async builtin. Per step: prefetch xw slab -> 60 paired
// WMMA tile jobs (A-fragment reused across 2 N-tiles) -> fused gate math.
// ---------------------------------------------------------------------------
__global__ __launch_bounds__(512)
void gru_layer_kernel(const float* __restrict__ xw,            // [ROWS, NPAD]
                      const unsigned short* __restrict__ WhhP, // [NPAD, HPAD] bf16
                      const float* __restrict__ bhh,           // [450]
                      const int* __restrict__ lengths,         // [64]
                      unsigned short* __restrict__ outBf16,    // [ROWS, HPAD] or null
                      float* __restrict__ outF32) {            // [ROWS, HPAD] or null
    extern __shared__ char smem[];
    unsigned short* sWhh = (unsigned short*)smem;                          // 153600
    float*          sHg  = (float*)(smem + NPAD * HPAD * 2);               // 122880
    unsigned short* sH   = (unsigned short*)(smem + NPAD * HPAD * 2 + BB * NPAD * 4);

    const int tid  = threadIdx.x;
    const int lane = tid & 31;
    const int wave = tid >> 5; // 0..15

    // ---- Whh -> LDS (async-to-LDS path when available) ----
#if __has_builtin(__builtin_amdgcn_global_load_async_to_lds_b128)
    {
        typedef int v4i __attribute__((vector_size(16)));   // b128 payload type
        __attribute__((address_space(1))) v4i* gsrc =
            (__attribute__((address_space(1))) v4i*)WhhP;
        __attribute__((address_space(3))) v4i* ldst =
            (__attribute__((address_space(3))) v4i*)sWhh;
        const int nChunks = NPAD * HPAD * 2 / 16;           // 9600 x 16B
        for (int i = tid; i < nChunks; i += 512)
            __builtin_amdgcn_global_load_async_to_lds_b128(gsrc + i, ldst + i, 0, 0);
#if __has_builtin(__builtin_amdgcn_s_wait_asynccnt)
        __builtin_amdgcn_s_wait_asynccnt(0);
#else
        asm volatile("s_wait_asynccnt 0" ::: "memory");
#endif
    }
#else
    for (int i = tid; i < NPAD * HPAD; i += 512) sWhh[i] = WhhP[i];
#endif
    for (int i = tid; i < BB * HPAD; i += 512) sH[i] = 0;
    __syncthreads();

    const int krow = (lane >> 4) << 4;

    for (int t = 0; t < TT; ++t) {
        const float* xwt = xw + (size_t)t * BB * NPAD;

        // Prefetch this step's input-projection slab (122,880 B / 256 B lines)
        {
            const char* p = (const char*)xwt;
            int off = tid * 256;
            if (off < BB * NPAD * 4) __builtin_prefetch(p + off, 0, 1);
        }

        // hg = h[64x160] x Whh^T ([N][K] layout) -> 4 M-tiles x 15 N-pairs
        for (int job = wave; job < 4 * (NPAD / 32); job += 16) {
            int mtile = job / (NPAD / 32);
            int npair = job - mtile * (NPAD / 32);
            const unsigned short* ar  = sH   + (mtile * 16 + (lane & 15)) * HPAD;
            const unsigned short* br0 = sWhh + (npair * 32 + (lane & 15)) * HPAD;
            const unsigned short* br1 = br0 + 16 * HPAD;
            v8f acc0 = {}, acc1 = {};
            for (int k = 0; k < HPAD; k += 32) {
                v16bf av  = *reinterpret_cast<const v16bf*>(ar  + k + krow);
                v16bf bv0 = *reinterpret_cast<const v16bf*>(br0 + k + krow);
                v16bf bv1 = *reinterpret_cast<const v16bf*>(br1 + k + krow);
                acc0 = wmma_bf16(av, bv0, acc0);
                acc1 = wmma_bf16(av, bv1, acc1);
            }
            int n0 = npair * 32 + (lane & 15);
            int n1 = n0 + 16;
            float bv0s = (n0 < H3) ? bhh[n0] : 0.0f;
            float bv1s = (n1 < H3) ? bhh[n1] : 0.0f;
            int mbase = mtile * 16 + ((lane >> 4) << 3);
            for (int i = 0; i < 8; ++i) {
                sHg[(mbase + i) * NPAD + n0] = acc0[i] + bv0s;
                sHg[(mbase + i) * NPAD + n1] = acc1[i] + bv1s;
            }
        }
        __syncthreads();

        // Fused gate math + masking + h update + output store
        for (int e = tid; e < BB * HPAD; e += 512) {
            int b = e / HPAD, j = e - b * HPAD;
            float hprev = bf16f(sH[e]);
            float outv = 0.0f, hnext = hprev;
            if (j < HH) {
                const float* row = xwt + b * NPAD;
                const float* hg  = sHg + b * NPAD;
                float r  = sigmoidf_(row[j]      + hg[j]);
                float z  = sigmoidf_(row[HH + j] + hg[HH + j]);
                float ng = tanhf(row[2 * HH + j] + r * hg[2 * HH + j]);
                float hnew = (1.0f - z) * ng + z * hprev;
                bool m = t < lengths[b];
                hnext = m ? hnew : hprev;
                outv  = m ? hnew : 0.0f;
            }
            size_t orow = ((size_t)t * BB + b) * HPAD + j;
            if (outBf16) outBf16[orow] = f32bf(outv);
            if (outF32)  outF32[orow]  = outv;
            sH[e] = f32bf(hnext);
        }
        __syncthreads();
    }
}

// ---------------------------------------------------------------------------
// Attention scores: scores[b][t] = sum_a w2[a] * tanh(out1[t,b,:] . W1[a,:])
// ---------------------------------------------------------------------------
__global__ void attn_scores(const float* __restrict__ out1,  // [ROWS, HPAD]
                            const float* __restrict__ W1,    // [75,150]
                            const float* __restrict__ w2,    // [75]
                            float* __restrict__ scores) {    // [B, T]
    int row = blockIdx.x * blockDim.x + threadIdx.x;
    if (row >= ROWS) return;
    int t = row >> 6, b = row & 63;
    const float* h = out1 + (size_t)row * HPAD;
    float s = 0.0f;
    for (int a = 0; a < 75; ++a) {
        const float* wr = W1 + a * HH;
        float d = 0.0f;
        for (int j = 0; j < HH; ++j) d += h[j] * wr[j];
        s += w2[a] * tanhf(d);
    }
    scores[b * TT + t] = s;
}

// ---------------------------------------------------------------------------
// Per-batch softmax over T, context vector, fc1+relu, fc2 -> logits[b]
// One block (256 thr) per batch element.
// ---------------------------------------------------------------------------
__global__ __launch_bounds__(256)
void softmax_ctx_head(const float* __restrict__ scores, // [B, T]
                      const float* __restrict__ out1,   // [ROWS, HPAD]
                      const float* __restrict__ fc1_w,  // [150,150]
                      const float* __restrict__ fc1_b,  // [150]
                      const float* __restrict__ fc2_w,  // [150]
                      const float* __restrict__ fc2_b,  // [1]
                      float* __restrict__ outLogits) {  // [B]
    __shared__ float sExp[TT];
    __shared__ float sRed[256];
    __shared__ float sCtx[HPAD];
    __shared__ float sHc[HPAD];
    int b = blockIdx.x;
    int tid = threadIdx.x;
    const float* srow = scores + b * TT;

    // max reduce
    float m = fmaxf(srow[tid], srow[tid + 256]);
    sRed[tid] = m;
    __syncthreads();
    for (int s = 128; s > 0; s >>= 1) {
        if (tid < s) sRed[tid] = fmaxf(sRed[tid], sRed[tid + s]);
        __syncthreads();
    }
    float mx = sRed[0];
    __syncthreads();

    // exp + sum
    float e0 = __expf(srow[tid] - mx);
    float e1 = __expf(srow[tid + 256] - mx);
    sExp[tid] = e0;
    sExp[tid + 256] = e1;
    sRed[tid] = e0 + e1;
    __syncthreads();
    for (int s = 128; s > 0; s >>= 1) {
        if (tid < s) sRed[tid] += sRed[tid + s];
        __syncthreads();
    }
    float inv = 1.0f / sRed[0];
    __syncthreads();

    // context: ctx[h] = (1/Z) sum_t exp[t] * out1[t,b,h]
    if (tid < HPAD) {
        float acc = 0.0f;
        if (tid < HH) {
            for (int t = 0; t < TT; ++t)
                acc += sExp[t] * out1[((size_t)t * BB + b) * HPAD + tid];
            acc *= inv;
        }
        sCtx[tid] = acc;
    }
    __syncthreads();

    // fc1 + relu
    if (tid < HH) {
        const float* wr = fc1_w + tid * HH;
        float acc = fc1_b[tid];
        for (int k = 0; k < HH; ++k) acc += wr[k] * sCtx[k];
        sHc[tid] = fmaxf(acc, 0.0f);
    }
    __syncthreads();

    // fc2
    if (tid == 0) {
        float acc = fc2_b[0];
        for (int j = 0; j < HH; ++j) acc += fc2_w[j] * sHc[j];
        outLogits[b] = acc;
    }
}

// ---------------------------------------------------------------------------
// Host orchestration
// ---------------------------------------------------------------------------
extern "C" void kernel_launch(void* const* d_in, const int* in_sizes, int n_in,
                              void* d_out, int out_size, void* d_ws, size_t ws_size,
                              hipStream_t stream) {
    const int*   texts   = (const int*)d_in[0];
    const int*   lengths = (const int*)d_in[1];
    const float* emb     = (const float*)d_in[2];
    const float* Wih0    = (const float*)d_in[3];
    const float* Whh0    = (const float*)d_in[4];
    const float* bih0    = (const float*)d_in[5];
    const float* bhh0    = (const float*)d_in[6];
    const float* Wih1    = (const float*)d_in[7];
    const float* Whh1    = (const float*)d_in[8];
    const float* bih1    = (const float*)d_in[9];
    const float* bhh1    = (const float*)d_in[10];
    const float* W1      = (const float*)d_in[11];
    const float* w2      = (const float*)d_in[12];
    const float* fc1_w   = (const float*)d_in[13];
    const float* fc1_b   = (const float*)d_in[14];
    const float* fc2_w   = (const float*)d_in[15];
    const float* fc2_b   = (const float*)d_in[16];
    float* logits = (float*)d_out;

    // Workspace partition (bytes, all 256B-aligned sizes)
    char* ws = (char*)d_ws;
    size_t off = 0;
    unsigned short* x0     = (unsigned short*)(ws + off); off += (size_t)ROWS * EPAD * 2;  // 20.97 MB
    unsigned short* Wih0p  = (unsigned short*)(ws + off); off += (size_t)NPAD * EPAD * 2;  // 307 KB
    unsigned short* Whh0p  = (unsigned short*)(ws + off); off += (size_t)NPAD * HPAD * 2;  // 153.6 KB
    unsigned short* Wih1p  = (unsigned short*)(ws + off); off += (size_t)NPAD * HPAD * 2;
    unsigned short* Whh1p  = (unsigned short*)(ws + off); off += (size_t)NPAD * HPAD * 2;
    float*          xw     = (float*)(ws + off);          off += (size_t)ROWS * NPAD * 4;  // 62.9 MB (reused L0/L1)
    unsigned short* out0b  = (unsigned short*)(ws + off); off += (size_t)ROWS * HPAD * 2;  // 10.5 MB
    float*          out1   = (float*)(ws + off);          off += (size_t)ROWS * HPAD * 4;  // 21 MB
    float*          scores = (float*)(ws + off);          off += (size_t)BB * TT * 4;      // 131 KB
    (void)ws_size; (void)in_sizes; (void)n_in; (void)out_size;

    // 1) Weight prep (fp32 -> padded bf16 [N][K])
    pad_convert<<<(NPAD * EPAD + 255) / 256, 256, 0, stream>>>(Wih0, H3, EE, Wih0p, NPAD, EPAD);
    pad_convert<<<(NPAD * HPAD + 255) / 256, 256, 0, stream>>>(Whh0, H3, HH, Whh0p, NPAD, HPAD);
    pad_convert<<<(NPAD * HPAD + 255) / 256, 256, 0, stream>>>(Wih1, H3, HH, Wih1p, NPAD, HPAD);
    pad_convert<<<(NPAD * HPAD + 255) / 256, 256, 0, stream>>>(Whh1, H3, HH, Whh1p, NPAD, HPAD);

    // 2) Embedding gather -> bf16
    {
        size_t total = (size_t)ROWS * EPAD;
        gather_embed<<<(unsigned)((total + 255) / 256), 256, 0, stream>>>(texts, emb, x0);
    }

    // 3) Layer-0 input projection: xw = x0 @ Wih0^T + bih0   (M=32768,K=320,N=480)
    {
        int jobs = (ROWS / 32) * (NPAD / 32);   // 15360 wave-jobs
        wmma_gemm_bf16<<<jobs / 8, 256, 0, stream>>>(x0, Wih0p, bih0, H3, xw, ROWS, EPAD, NPAD);
    }

    // 4) Layer-0 recurrence (persistent WGP-resident kernel)
    gru_layer_kernel<<<1, 512, 296960, stream>>>(xw, Whh0p, bhh0, lengths, out0b, nullptr);

    // 5) Layer-1 input projection: xw = out0 @ Wih1^T + bih1  (K=160)
    {
        int jobs = (ROWS / 32) * (NPAD / 32);
        wmma_gemm_bf16<<<jobs / 8, 256, 0, stream>>>(out0b, Wih1p, bih1, H3, xw, ROWS, HPAD, NPAD);
    }

    // 6) Layer-1 recurrence
    gru_layer_kernel<<<1, 512, 296960, stream>>>(xw, Whh1p, bhh1, lengths, nullptr, out1);

    // 7) Attention scores
    attn_scores<<<ROWS / 256, 256, 0, stream>>>(out1, W1, w2, scores);

    // 8) Softmax + context + classifier head
    softmax_ctx_head<<<BB, 256, 0, stream>>>(scores, out1, fc1_w, fc1_b, fc2_w, fc2_b, logits);
}
